// KarpathyTransformerModel_13073880449830
// MI455X (gfx1250) — matrix operand order repaired and checked by reading, not compiled, not run on hardware
//
#include <hip/hip_runtime.h>

typedef __attribute__((ext_vector_type(16))) __bf16 v16bf;
typedef __attribute__((ext_vector_type(8)))  float  v8f;
typedef unsigned short ushort_t;
typedef unsigned int   uint_t;

// ---------------------------------------------------------------- helpers ---
__device__ __forceinline__ ushort_t f2bf(float f) {
  unsigned int u = __float_as_uint(f);
  u += 0x7FFFu + ((u >> 16) & 1u);     // round-to-nearest-even
  return (ushort_t)(u >> 16);
}
__device__ __forceinline__ uint_t pack2bf(float a, float b) {
  return (uint_t)f2bf(a) | ((uint_t)f2bf(b) << 16);
}

// A fragment (16x32 bf16) from LDS tile, row stride 32 (ISA 7.12.2 16-bit A):
// lanes 0-15: M=lane, K = {0..7, 16..23}; lanes 16-31: M=lane-16, K = {8..15, 24..31}
__device__ __forceinline__ v16bf frag_a_lds(const ushort_t* base, int lane) {
  union { int i[8]; v16bf v; } f;
  const int half = (lane >> 4) << 3;
  const ushort_t* p = base + (lane & 15) * 32;
#pragma unroll
  for (int r = 0; r < 8; ++r) {
    int k = (r < 4 ? 2 * r : 2 * r + 8) + half;
    f.i[r] = *(const int*)(p + k);
  }
  return f.v;
}

// B fragment (32x16 bf16) from transposed LDS tile sBt[col*32+k]:
// lanes 0-15: N=lane, K = {0..7, 16..23}; lanes 16-31: N=lane-16, K = {8..15, 24..31} (pairs per VGPR)
__device__ __forceinline__ v16bf frag_b_lds(const ushort_t* baseT, int lane) {
  union { int i[8]; v16bf v; } f;
  const int kh = (lane >> 4) << 3;
  const ushort_t* p = baseT + (lane & 15) * 32 + kh;
#pragma unroll
  for (int r = 0; r < 8; ++r) f.i[r] = *(const int*)(p + (r < 4 ? 2 * r : 2 * r + 8));
  return f.v;
}

#define WMMA_BF16(a, b, c) \
  __builtin_amdgcn_wmma_f32_16x16x32_bf16(false, (a), false, (b), (short)0, (c), false, false)

// ------------------------------------------------------ generic WMMA GEMM ---
// C[M,N] (f32) = A(bf16, MxK) @ B(bf16, KxN) + bias.
// WG tile 128x128, 8 waves (4x2), wave tile 32x64 (2x4 frags), K-step 32.
// Register-double-buffered global->LDS pipeline.
__global__ __launch_bounds__(256) void k_wmma_gemm_bf16(
    const ushort_t* __restrict__ A, const ushort_t* __restrict__ B,
    const float* __restrict__ bias, float* __restrict__ C, int M, int N, int K) {
  __shared__ alignas(16) ushort_t sA[128 * 32];
  __shared__ alignas(16) ushort_t sB[128 * 32];   // transposed: [col*32 + k]
  const int tid = threadIdx.x, lane = tid & 31, wave = tid >> 5;
  const int wm = (wave >> 1) * 32, wn = (wave & 1) * 64;
  const int m0 = blockIdx.y * 128, n0 = blockIdx.x * 128;
  const v8f vz = {0.f, 0.f, 0.f, 0.f, 0.f, 0.f, 0.f, 0.f};
  v8f acc[2][4];
#pragma unroll
  for (int mi = 0; mi < 2; ++mi)
#pragma unroll
    for (int ni = 0; ni < 4; ++ni) acc[mi][ni] = vz;
  const int arow = tid >> 1, a4o = (tid & 1) * 2;   // A: 16 contig ushorts / thread
  int bcol[4], bkk[4];                               // B: 4x uint2 (4 cols) / thread
#pragma unroll
  for (int g = 0; g < 4; ++g) {
    int gidx = g * 256 + tid;
    bcol[g] = (gidx & 31) * 4;
    bkk[g] = gidx >> 5;
  }
  uint4* sA4 = (uint4*)sA;
  uint4 ra0, ra1; uint2 rb[4];
  auto load_tiles = [&](int k0_) {
    const uint4* Ag = (const uint4*)(A + (size_t)(m0 + arow) * K + k0_);
    ra0 = Ag[a4o]; ra1 = Ag[a4o + 1];
#pragma unroll
    for (int g = 0; g < 4; ++g)
      rb[g] = *(const uint2*)(B + (size_t)(k0_ + bkk[g]) * N + (n0 + bcol[g]));
  };
  load_tiles(0);
  for (int k0 = 0; k0 < K; k0 += 32) {
    __syncthreads();
    sA4[arow * 4 + a4o] = ra0; sA4[arow * 4 + a4o + 1] = ra1;
#pragma unroll
    for (int g = 0; g < 4; ++g) {
      int c = bcol[g], kk = bkk[g];
      sB[(c + 0) * 32 + kk] = (ushort_t)rb[g].x;
      sB[(c + 1) * 32 + kk] = (ushort_t)(rb[g].x >> 16);
      sB[(c + 2) * 32 + kk] = (ushort_t)rb[g].y;
      sB[(c + 3) * 32 + kk] = (ushort_t)(rb[g].y >> 16);
    }
    int kn = (k0 + 32 < K) ? (k0 + 32) : k0;
    load_tiles(kn);                                  // next tile, latency hidden
    if (k0 + 64 < K)
      __builtin_prefetch(B + (size_t)(k0 + 64 + (tid >> 6)) * N + n0 + (tid & 63), 0, 1);
    __syncthreads();
    v16bf a0 = frag_a_lds(sA + wm * 32, lane);
    v16bf a1 = frag_a_lds(sA + (wm + 16) * 32, lane);
#pragma unroll
    for (int ni = 0; ni < 4; ++ni) {
      v16bf bf = frag_b_lds(sB + (wn + ni * 16) * 32, lane);
      acc[0][ni] = WMMA_BF16(a0, bf, acc[0][ni]);
      acc[1][ni] = WMMA_BF16(a1, bf, acc[1][ni]);
    }
  }
  const int half = lane >> 4, nl = lane & 15;
#pragma unroll
  for (int mi = 0; mi < 2; ++mi) {
    int row0 = m0 + wm + mi * 16 + half * 8;
#pragma unroll
    for (int ni = 0; ni < 4; ++ni) {
      int col = n0 + wn + ni * 16 + nl;
      float bv = bias[col];
      float* Cp = C + (size_t)row0 * N + col;
#pragma unroll
      for (int r = 0; r < 8; ++r) { *Cp = acc[mi][ni][r] + bv; Cp += N; }
    }
  }
}

// -------------------------------------------- fused pair-MLP layer-1 GEMM ---
// H2[m, n] = bf16( relu(qp[b,i,:]+kp[b,j,:]+b1) @ W2 + b2 ), m=((b*128+i)*128+j)
// A tile (h1) constructed on the fly; raw f32 loads pipelined. M=32768, N=K=1024.
__global__ __launch_bounds__(256) void k_pair_gemm1(
    const float* __restrict__ qp, const float* __restrict__ kp,
    const float* __restrict__ b1, const ushort_t* __restrict__ W2,
    const float* __restrict__ b2, ushort_t* __restrict__ H2) {
  __shared__ alignas(16) ushort_t sA[128 * 32];
  __shared__ alignas(16) ushort_t sB[128 * 32];
  const int tid = threadIdx.x, lane = tid & 31, wave = tid >> 5;
  const int wm = (wave >> 1) * 32, wn = (wave & 1) * 64;
  const int m0 = blockIdx.y * 128, n0 = blockIdx.x * 128;
  const v8f vz = {0.f, 0.f, 0.f, 0.f, 0.f, 0.f, 0.f, 0.f};
  v8f acc[2][4];
#pragma unroll
  for (int mi = 0; mi < 2; ++mi)
#pragma unroll
    for (int ni = 0; ni < 4; ++ni) acc[mi][ni] = vz;
  const int arow = tid >> 1, koff = (tid & 1) * 16;
  const int m = m0 + arow;
  const int qrow = m >> 7;                          // b*T + i
  const int krow = ((m >> 14) << 7) | (m & 127);    // b*T + j
  int bcol[4], bkk[4];
#pragma unroll
  for (int g = 0; g < 4; ++g) {
    int gidx = g * 256 + tid;
    bcol[g] = (gidx & 31) * 4;
    bkk[g] = gidx >> 5;
  }
  uint4* sA4 = (uint4*)sA;
  float4 rq[4], rk[4], rv[4]; uint2 rb[4];
  auto load_tiles = [&](int k0_) {
    const float4* q4 = (const float4*)(qp + (size_t)qrow * 1024 + k0_ + koff);
    const float4* k4 = (const float4*)(kp + (size_t)krow * 1024 + k0_ + koff);
    const float4* b4 = (const float4*)(b1 + k0_ + koff);
#pragma unroll
    for (int e = 0; e < 4; ++e) { rq[e] = q4[e]; rk[e] = k4[e]; rv[e] = b4[e]; }
#pragma unroll
    for (int g = 0; g < 4; ++g)
      rb[g] = *(const uint2*)(W2 + (size_t)(k0_ + bkk[g]) * 1024 + (n0 + bcol[g]));
  };
  load_tiles(0);
  for (int k0 = 0; k0 < 1024; k0 += 32) {
    __syncthreads();
    {  // pack relu(q+k+b1) -> bf16 tile
      uint_t o[8];
#pragma unroll
      for (int e = 0; e < 4; ++e) {
        o[e * 2]     = pack2bf(fmaxf(rq[e].x + rk[e].x + rv[e].x, 0.f),
                               fmaxf(rq[e].y + rk[e].y + rv[e].y, 0.f));
        o[e * 2 + 1] = pack2bf(fmaxf(rq[e].z + rk[e].z + rv[e].z, 0.f),
                               fmaxf(rq[e].w + rk[e].w + rv[e].w, 0.f));
      }
      uint4 w0; w0.x = o[0]; w0.y = o[1]; w0.z = o[2]; w0.w = o[3];
      uint4 w1; w1.x = o[4]; w1.y = o[5]; w1.z = o[6]; w1.w = o[7];
      int bidx = arow * 4 + (tid & 1) * 2;
      sA4[bidx] = w0; sA4[bidx + 1] = w1;
    }
#pragma unroll
    for (int g = 0; g < 4; ++g) {
      int c = bcol[g], kk = bkk[g];
      sB[(c + 0) * 32 + kk] = (ushort_t)rb[g].x;
      sB[(c + 1) * 32 + kk] = (ushort_t)(rb[g].x >> 16);
      sB[(c + 2) * 32 + kk] = (ushort_t)rb[g].y;
      sB[(c + 3) * 32 + kk] = (ushort_t)(rb[g].y >> 16);
    }
    int kn = (k0 + 32 < 1024) ? (k0 + 32) : k0;
    load_tiles(kn);
    if (k0 + 64 < 1024)
      __builtin_prefetch(W2 + (size_t)(k0 + 64 + (tid >> 6)) * 1024 + n0 + (tid & 63), 0, 1);
    __syncthreads();
    v16bf a0 = frag_a_lds(sA + wm * 32, lane);
    v16bf a1 = frag_a_lds(sA + (wm + 16) * 32, lane);
#pragma unroll
    for (int ni = 0; ni < 4; ++ni) {
      v16bf bf = frag_b_lds(sB + (wn + ni * 16) * 32, lane);
      acc[0][ni] = WMMA_BF16(a0, bf, acc[0][ni]);
      acc[1][ni] = WMMA_BF16(a1, bf, acc[1][ni]);
    }
  }
  const int half = lane >> 4, nl = lane & 15;
#pragma unroll
  for (int mi = 0; mi < 2; ++mi) {
    int row0 = m0 + wm + mi * 16 + half * 8;
#pragma unroll
    for (int ni = 0; ni < 4; ++ni) {
      int col = n0 + wn + ni * 16 + nl;
      float bv = b2[col];
      ushort_t* Hp = H2 + (size_t)row0 * 1024 + col;
#pragma unroll
      for (int r = 0; r < 8; ++r) { *Hp = f2bf(acc[mi][ni][r] + bv); Hp += 1024; }
    }
  }
}

// ----------------------- pair-MLP layer-2 GEMM + fused W4/att2 reduction ---
// part[slot, m] = sum_{n in slot's 64 cols} relu(H2@W3 + b3)[m,n] * w4a2[n]
// (deterministic partial sums; 16 slots cover N=1024)
__global__ __launch_bounds__(256) void k_pair_gemm2(
    const ushort_t* __restrict__ H2, const ushort_t* __restrict__ W3,
    const float* __restrict__ b3, const float* __restrict__ w4a2,
    float* __restrict__ part) {
  __shared__ alignas(16) ushort_t sA[128 * 32];
  __shared__ alignas(16) ushort_t sB[128 * 32];
  const int tid = threadIdx.x, lane = tid & 31, wave = tid >> 5;
  const int wm = (wave >> 1) * 32, wn = (wave & 1) * 64;
  const int m0 = blockIdx.y * 128, n0 = blockIdx.x * 128;
  const v8f vz = {0.f, 0.f, 0.f, 0.f, 0.f, 0.f, 0.f, 0.f};
  v8f acc[2][4];
#pragma unroll
  for (int mi = 0; mi < 2; ++mi)
#pragma unroll
    for (int ni = 0; ni < 4; ++ni) acc[mi][ni] = vz;
  const int arow = tid >> 1, a4o = (tid & 1) * 2;
  int bcol[4], bkk[4];
#pragma unroll
  for (int g = 0; g < 4; ++g) {
    int gidx = g * 256 + tid;
    bcol[g] = (gidx & 31) * 4;
    bkk[g] = gidx >> 5;
  }
  uint4* sA4 = (uint4*)sA;
  uint4 ra0, ra1; uint2 rb[4];
  auto load_tiles = [&](int k0_) {
    const uint4* Ag = (const uint4*)(H2 + (size_t)(m0 + arow) * 1024 + k0_);
    ra0 = Ag[a4o]; ra1 = Ag[a4o + 1];
#pragma unroll
    for (int g = 0; g < 4; ++g)
      rb[g] = *(const uint2*)(W3 + (size_t)(k0_ + bkk[g]) * 1024 + (n0 + bcol[g]));
  };
  load_tiles(0);
  for (int k0 = 0; k0 < 1024; k0 += 32) {
    __syncthreads();
    sA4[arow * 4 + a4o] = ra0; sA4[arow * 4 + a4o + 1] = ra1;
#pragma unroll
    for (int g = 0; g < 4; ++g) {
      int c = bcol[g], kk = bkk[g];
      sB[(c + 0) * 32 + kk] = (ushort_t)rb[g].x;
      sB[(c + 1) * 32 + kk] = (ushort_t)(rb[g].x >> 16);
      sB[(c + 2) * 32 + kk] = (ushort_t)rb[g].y;
      sB[(c + 3) * 32 + kk] = (ushort_t)(rb[g].y >> 16);
    }
    int kn = (k0 + 32 < 1024) ? (k0 + 32) : k0;
    load_tiles(kn);
    if (k0 + 64 < 1024)
      __builtin_prefetch(W3 + (size_t)(k0 + 64 + (tid >> 6)) * 1024 + n0 + (tid & 63), 0, 1);
    __syncthreads();
    v16bf a0 = frag_a_lds(sA + wm * 32, lane);
    v16bf a1 = frag_a_lds(sA + (wm + 16) * 32, lane);
#pragma unroll
    for (int ni = 0; ni < 4; ++ni) {
      v16bf bf = frag_b_lds(sB + (wn + ni * 16) * 32, lane);
      acc[0][ni] = WMMA_BF16(a0, bf, acc[0][ni]);
      acc[1][ni] = WMMA_BF16(a1, bf, acc[1][ni]);
    }
  }
  const int half = lane >> 4, nl = lane & 15;
  float b3v[4], wv[4];
#pragma unroll
  for (int ni = 0; ni < 4; ++ni) {
    int col = n0 + wn + ni * 16 + nl;
    b3v[ni] = b3[col];
    wv[ni] = w4a2[col];
  }
#pragma unroll
  for (int mi = 0; mi < 2; ++mi)
#pragma unroll
    for (int r = 0; r < 8; ++r) {
      float p = 0.f;
#pragma unroll
      for (int ni = 0; ni < 4; ++ni)
        p += fmaxf(acc[mi][ni][r] + b3v[ni], 0.f) * wv[ni];
      p += __shfl_xor(p, 1); p += __shfl_xor(p, 2);
      p += __shfl_xor(p, 4); p += __shfl_xor(p, 8);   // reduce within 16-lane half
      if (nl == 0) {
        int slot = blockIdx.x * 2 + (wn >> 6);        // 16 N-slots of 64 cols
        int row = m0 + wm + mi * 16 + r + half * 8;
        part[(size_t)slot * 32768 + row] = p;
      }
    }
}

__global__ void k_reduce_wei(const float* __restrict__ part, float* __restrict__ wei) {
  int m = blockIdx.x * 256 + threadIdx.x;
  float s = 0.f;
#pragma unroll
  for (int p = 0; p < 16; ++p) s += part[p * 32768 + m];
  wei[m] = s;
}

// ------------------------------------------------------------ small kernels -
__global__ void k_gemm_naive(const float* __restrict__ A, int lda,
                             const float* __restrict__ B, int ldb,
                             const float* __restrict__ bias,
                             const float* __restrict__ add,
                             float* __restrict__ C, int ldc,
                             int M, int N, int K, int relu) {
  int n = blockIdx.x * blockDim.x + threadIdx.x;
  int m = blockIdx.y;
  if (n >= N || m >= M) return;
  float acc = bias ? bias[n] : 0.f;
  for (int k = 0; k < K; ++k) acc += A[(size_t)m * lda + k] * B[(size_t)k * ldb + n];
  if (relu) acc = fmaxf(acc, 0.f);
  if (add) acc += add[(size_t)m * ldc + n];
  C[(size_t)m * ldc + n] = acc;
}

__global__ void k_layernorm(const float* __restrict__ x, const float* __restrict__ g,
                            const float* __restrict__ b, float* __restrict__ y) {
  int row = blockIdx.x, t = threadIdx.x;                 // C = 256, block 256
  float v = x[row * 256 + t];
  __shared__ float s1[8], s2[8];
  float a = v, q = v * v;
  for (int o = 16; o; o >>= 1) { a += __shfl_xor(a, o); q += __shfl_xor(q, o); }
  if ((t & 31) == 0) { s1[t >> 5] = a; s2[t >> 5] = q; }
  __syncthreads();
  float sum = 0.f, sq = 0.f;
#pragma unroll
  for (int w = 0; w < 8; ++w) { sum += s1[w]; sq += s2[w]; }
  float mean = sum * (1.f / 256.f);
  float var = sq * (1.f / 256.f) - mean * mean;
  y[row * 256 + t] = (v - mean) * rsqrtf(var + 1e-5f) * g[t] + b[t];
}

__global__ void k_gather(const float* __restrict__ tok, const int* __restrict__ idx,
                         float* __restrict__ x) {
  int bt = blockIdx.x, c = threadIdx.x;
  x[bt * 256 + c] = tok[(size_t)idx[bt] * 256 + c];
}

__global__ void k_concat_st(const float* __restrict__ pos, const float* __restrict__ step,
                            float* __restrict__ o) {        // (128,512)
  int t = blockIdx.x, c = threadIdx.x;
  o[t * 512 + c] = (c < 256) ? pos[t * 256 + c] : step[c - 256];
}

__global__ void k_concat_x1(const float* __restrict__ sp, const float* __restrict__ xn,
                            float* __restrict__ o) {        // (256,512)
  int bt = blockIdx.x, c = threadIdx.x;
  int t = bt & 127;
  o[bt * 512 + c] = (c < 256) ? sp[t * 256 + c] : xn[bt * 256 + (c - 256)];
}

__global__ void k_f32_to_bf16(const float* __restrict__ s, ushort_t* __restrict__ d, int n) {
  int i = blockIdx.x * blockDim.x + threadIdx.x;
  if (i < n) d[i] = f2bf(s[i]);
}

__global__ void k_fold_w4(const float* __restrict__ W4, const float* __restrict__ a2,
                          float* __restrict__ o) {          // o[k]=W4[k,:]·a2, k<1024
  int k = blockIdx.x * blockDim.x + threadIdx.x;
  float s = 0.f;
  for (int n = 0; n < 256; ++n) s += W4[k * 256 + n] * a2[n];
  o[k] = s;
}

__global__ void k_softmax_causal(float* __restrict__ wei, float scale) {
  int row = blockIdx.x;                                     // b*T + i ; block 128
  int i = row & 127, j = threadIdx.x;
  float* w = wei + row * 128;
  bool valid = (j <= i);
  float v = valid ? w[j] * scale : -3.0e38f;
  __shared__ float rmax[4], rsum[4];
  float m = v;
  for (int o = 16; o; o >>= 1) m = fmaxf(m, __shfl_xor(m, o));
  if ((j & 31) == 0) rmax[j >> 5] = m;
  __syncthreads();
  m = fmaxf(fmaxf(rmax[0], rmax[1]), fmaxf(rmax[2], rmax[3]));
  float e = valid ? __expf(v - m) : 0.f;
  float s = e;
  for (int o = 16; o; o >>= 1) s += __shfl_xor(s, o);
  if ((j & 31) == 0) rsum[j >> 5] = s;
  __syncthreads();
  s = rsum[0] + rsum[1] + rsum[2] + rsum[3];
  w[j] = e / s;
}

// ------------------------------------------------------------------- host ---
static void gemm_naive(hipStream_t s, const float* A, int lda, const float* B, int ldb,
                       const float* bias, const float* add, float* C, int ldc,
                       int M, int N, int K, int relu) {
  dim3 g((N + 127) / 128, M);
  k_gemm_naive<<<g, 128, 0, s>>>(A, lda, B, ldb, bias, add, C, ldc, M, N, K, relu);
}

extern "C" void kernel_launch(void* const* d_in, const int* in_sizes, int n_in,
                              void* d_out, int out_size, void* d_ws, size_t ws_size,
                              hipStream_t stream) {
  (void)in_sizes; (void)n_in; (void)out_size; (void)ws_size;
  // input order = setup_inputs() dict order (idx, then params in insertion order)
  const int* idx = (const int*)d_in[0];
  auto F = [&](int i) { return (const float*)d_in[i]; };
  const float *tok = F(1),
    *peT = F(2), *peW1 = F(3), *peb1 = F(4), *peW2 = F(5), *peb2 = F(6),
    *seT = F(7), *seW1 = F(8), *seb1 = F(9), *seW2 = F(10), *seb2 = F(11),
    *stW1 = F(12), *stb1 = F(13), *stW2 = F(14), *stb2 = F(15),
    *attW1 = F(16), *attb1 = F(17), *attW2 = F(18), *attb2 = F(19),
    *attW3 = F(20), *attb3 = F(21), *attW4 = F(22), /* attb4 folded out */
    *blkW1 = F(24), *blkb1 = F(25), *blkW2 = F(26), *blkb2 = F(27),
    *ln1g = F(28), *ln1b = F(29),
    *hffW1 = F(30), *hffb1 = F(31), *hffW2 = F(32), *hffb2 = F(33),
    *att2W = F(34), /* att2b folded out (softmax shift-invariance) */
    *valW = F(36), *projW = F(37), *projb = F(38),
    *ln2g = F(39), *ln2b = F(40),
    *ffW1 = F(41), *ffb1 = F(42), *ffW2 = F(43), *ffb2 = F(44),
    *lnfg = F(45), *lnfb = F(46), *lmW = F(47), *lmb = F(48);

  // workspace carve
  char* base = (char*)d_ws;
  size_t off = 0;
  auto alloc = [&](size_t bytes) -> void* {
    void* p = base + off; off += (bytes + 255) & ~(size_t)255; return p;
  };
  float* pos_h  = (float*)alloc(128 * 256 * 4);
  float* pos    = (float*)alloc(128 * 256 * 4);
  float* step_h = (float*)alloc(256 * 4);
  float* step   = (float*)alloc(256 * 4);
  float* stcat  = (float*)alloc(128 * 512 * 4);
  float* st_h   = (float*)alloc(128 * 256 * 4);
  float* st0    = (float*)alloc(128 * 256 * 4);
  float* blk_h  = (float*)alloc(128 * 1024 * 4);
  float* st     = (float*)alloc(128 * 256 * 4);
  float* x      = (float*)alloc(256 * 256 * 4);
  float* xn     = (float*)alloc(256 * 256 * 4);
  float* sp_h   = (float*)alloc(128 * 256 * 4);
  float* sp     = (float*)alloc(128 * 256 * 4);
  float* x1     = (float*)alloc(256 * 512 * 4);
  float* kp     = (float*)alloc(256 * 1024 * 4);
  float* qp     = (float*)alloc(256 * 1024 * 4);
  float* w4a2   = (float*)alloc(1024 * 4);
  float* wei    = (float*)alloc(32768 * 4);
  float* wpart  = (float*)alloc((size_t)16 * 32768 * 4);
  float* vh     = (float*)alloc(256 * 128 * 4);
  float* headc  = (float*)alloc(256 * 256 * 4);
  float* x2     = (float*)alloc(256 * 256 * 4);
  float* t1     = (float*)alloc(256 * 256 * 4);
  float* t2     = (float*)alloc(256 * 1024 * 4);
  float* x3     = (float*)alloc(256 * 256 * 4);
  float* xf     = (float*)alloc(256 * 256 * 4);
  ushort_t* W2bf  = (ushort_t*)alloc((size_t)1024 * 1024 * 2);
  ushort_t* W3bf  = (ushort_t*)alloc((size_t)1024 * 1024 * 2);
  ushort_t* lmWbf = (ushort_t*)alloc((size_t)256 * 32000 * 2);
  ushort_t* xfbf  = (ushort_t*)alloc((size_t)256 * 256 * 2);
  ushort_t* H2    = (ushort_t*)alloc((size_t)32768 * 1024 * 2);

  // ---- embeddings / st path (batch-independent where possible)
  gemm_naive(stream, peT, 256, peW1, 256, peb1, nullptr, pos_h, 256, 128, 256, 256, 1);
  gemm_naive(stream, pos_h, 256, peW2, 256, peb2, nullptr, pos, 256, 128, 256, 256, 0);
  gemm_naive(stream, seT, 256, seW1, 256, seb1, nullptr, step_h, 256, 1, 256, 256, 1);
  gemm_naive(stream, step_h, 256, seW2, 256, seb2, nullptr, step, 256, 1, 256, 256, 0);
  k_concat_st<<<128, 512, 0, stream>>>(pos, step, stcat);
  gemm_naive(stream, stcat, 512, stW1, 256, stb1, nullptr, st_h, 256, 128, 256, 512, 1);
  gemm_naive(stream, st_h, 256, stW2, 256, stb2, nullptr, st0, 256, 128, 256, 256, 0);
  gemm_naive(stream, st0, 256, blkW1, 1024, blkb1, nullptr, blk_h, 1024, 128, 1024, 256, 1);
  gemm_naive(stream, blk_h, 1024, blkW2, 256, blkb2, st0, st, 256, 128, 256, 1024, 0);
  k_gather<<<256, 256, 0, stream>>>(tok, idx, x);
  k_layernorm<<<256, 256, 0, stream>>>(x, ln1g, ln1b, xn);

  // ---- one-time bf16 weight conversions
  k_f32_to_bf16<<<4096, 256, 0, stream>>>(attW2, W2bf, 1024 * 1024);
  k_f32_to_bf16<<<4096, 256, 0, stream>>>(attW3, W3bf, 1024 * 1024);
  k_f32_to_bf16<<<32000, 256, 0, stream>>>(lmW, lmWbf, 256 * 32000);

  // ---- heads
  for (int h = 0; h < 2; ++h) {
    gemm_naive(stream, st, 256, hffW1 + (size_t)h * 256 * 256, 256,
               hffb1 + h * 256, nullptr, sp_h, 256, 128, 256, 256, 1);
    gemm_naive(stream, sp_h, 256, hffW2 + (size_t)h * 256 * 256, 256,
               hffb2 + h * 256, nullptr, sp, 256, 128, 256, 256, 0);
    k_concat_x1<<<256, 512, 0, stream>>>(sp, xn, x1);
    gemm_naive(stream, x1, 512, attW1, 1024, nullptr, nullptr, kp, 1024, 256, 1024, 512, 0);
    gemm_naive(stream, x1, 512, attW1 + (size_t)512 * 1024, 1024, nullptr, nullptr,
               qp, 1024, 256, 1024, 512, 0);
    k_fold_w4<<<8, 128, 0, stream>>>(attW4, att2W + h * 256, w4a2);

    dim3 gp(8, 256);  // N/128 x M/128
    k_pair_gemm1<<<gp, 256, 0, stream>>>(qp, kp, attb1, W2bf, attb2, H2);
    k_pair_gemm2<<<gp, 256, 0, stream>>>(H2, W3bf, attb3, w4a2, wpart);
    k_reduce_wei<<<128, 256, 0, stream>>>(wpart, wei);
    k_softmax_causal<<<256, 128, 0, stream>>>(wei, 0.0625f);  // C^-0.5 = 1/16

    gemm_naive(stream, xn, 256, valW + (size_t)h * 256 * 128, 128, nullptr, nullptr,
               vh, 128, 256, 128, 256, 0);
    for (int b = 0; b < 2; ++b)
      gemm_naive(stream, wei + b * 128 * 128, 128, vh + b * 128 * 128, 128,
                 nullptr, nullptr, headc + b * 128 * 256 + h * 128, 256,
                 128, 128, 128, 0);
  }

  // ---- projection + residual + FF + final LN
  gemm_naive(stream, headc, 256, projW, 256, projb, x, x2, 256, 256, 256, 256, 0);
  k_layernorm<<<256, 256, 0, stream>>>(x2, ln2g, ln2b, t1);
  gemm_naive(stream, t1, 256, ffW1, 1024, ffb1, nullptr, t2, 1024, 256, 1024, 256, 1);
  gemm_naive(stream, t2, 1024, ffW2, 256, ffb2, x2, x3, 256, 256, 256, 1024, 0);
  k_layernorm<<<256, 256, 0, stream>>>(x3, lnfg, lnfb, xf);

  // ---- lm head (WMMA): logits (256 x 32000) f32 -> d_out
  k_f32_to_bf16<<<256, 256, 0, stream>>>(xf, xfbf, 256 * 256);
  dim3 gl(250, 2);
  k_wmma_gemm_bf16<<<gl, 256, 0, stream>>>(xfbf, lmWbf, lmb, (float*)d_out,
                                           256, 32000, 256);
}
